// ModulatedDeformConv_1726576855659
// MI455X (gfx1250) — compile-verified, matrix operand order checked by reading, not run on hardware
//
#include <hip/hip_runtime.h>

typedef _Float16 v16h __attribute__((ext_vector_type(16)));
typedef float    v8f  __attribute__((ext_vector_type(8)));

#define NB    2     // batch
#define CC    128   // in channels
#define HH    128
#define WW    128
#define COUT  128
#define KK    9     // 3x3 taps
#define CK    1152  // C*K (GEMM K dim)
#define TP    32    // pixel tile per workgroup (2 x 16 GEMM N tiles)
#define LDSS  (CK + 4)  // 1156 halves = 578 dw; 578%64=2 -> 16 pixel rows hit distinct banks

// ---------------- weight f32 -> f16 (one shot, into workspace) ----------------
__global__ __launch_bounds__(256) void dcn_w2h(const float* __restrict__ w,
                                               _Float16* __restrict__ wh, int n) {
  int i = blockIdx.x * 256 + threadIdx.x;
  if (i < n) wh[i] = (_Float16)w[i];
}

// ---------------- fused deformable-im2col + WMMA GEMM ----------------
__global__ __launch_bounds__(256) void dcn_main(const float* __restrict__ x,
                                                const float* __restrict__ offset,
                                                const float* __restrict__ mask,
                                                const _Float16* __restrict__ wh,
                                                const float* __restrict__ bias,
                                                float* __restrict__ out) {
  __shared__ _Float16 cols[TP * LDSS];            // cols[pixel][ck]  (f16), ~72 KB
  __shared__ float sY[KK * TP], sX[KK * TP], sM[KK * TP];

  const int tid  = threadIdx.x;
  const int tile = blockIdx.x;                    // 1024 tiles; 512 per batch image
  const int nb   = tile >> 9;
  const int pixBase = (tile & 511) << 5;          // linear ho*WO+wo base (never crosses batch)

  // ---- stage sampling coordinates + mask for 9 taps x 32 pixels (288 entries) ----
  for (int i = tid; i < KK * TP; i += 256) {
    const int k = i >> 5, p = i & 31;
    const int howo = pixBase + p;
    const int ho = howo >> 7, wo = howo & 127;
    const float offy = offset[(((nb * 2 * KK) + 2 * k)     << 14) + howo];
    const float offx = offset[(((nb * 2 * KK) + 2 * k + 1) << 14) + howo];
    sY[i] = (float)(ho - 1 + (k / 3)) + offy;
    sX[i] = (float)(wo - 1 + (k % 3)) + offx;
    sM[i] = mask[((nb * KK + k) << 14) + howo];
  }
  __syncthreads();

  // ---- deformable im2col: 1152 x 32 tile, f32 bilinear sample -> f16 LDS ----
  for (int idx = tid; idx < CK * TP; idx += 256) {
    const int p  = idx & 31;                      // consecutive lanes -> consecutive pixels
    const int ck = idx >> 5;
    const int c  = ck / 9;
    const int k  = ck - 9 * c;
    const float ys = sY[(k << 5) + p];
    const float xs = sX[(k << 5) + p];
    const float mm = sM[(k << 5) + p];
    const float* __restrict__ img = x + (((nb * CC) + c) << 14);

    const float y0f = floorf(ys), x0f = floorf(xs);
    const float ly = ys - y0f, lx = xs - x0f;
    const int y0 = (int)y0f, x0 = (int)x0f;
    const int y1 = y0 + 1,  x1 = x0 + 1;
    const bool vy0 = (y0 >= 0) & (y0 < HH);
    const bool vy1 = (y1 >= 0) & (y1 < HH);
    const bool vx0 = (x0 >= 0) & (x0 < WW);
    const bool vx1 = (x1 >= 0) & (x1 < WW);
    const int y0c = min(max(y0, 0), HH - 1), y1c = min(max(y1, 0), HH - 1);
    const int x0c = min(max(x0, 0), WW - 1), x1c = min(max(x1, 0), WW - 1);
    const float v00 = (vy0 && vx0) ? img[(y0c << 7) + x0c] : 0.f;
    const float v01 = (vy0 && vx1) ? img[(y0c << 7) + x1c] : 0.f;
    const float v10 = (vy1 && vx0) ? img[(y1c << 7) + x0c] : 0.f;
    const float v11 = (vy1 && vx1) ? img[(y1c << 7) + x1c] : 0.f;
    const float val = v00 * (1.f - ly) * (1.f - lx) + v01 * (1.f - ly) * lx +
                      v10 * ly * (1.f - lx)         + v11 * ly * lx;
    cols[p * LDSS + ck] = (_Float16)(val * mm);
  }
  __syncthreads();

  // ---- GEMM: wave w -> COUT rows [16w,16w+16) x 2 pixel sub-tiles of 16 ----
  const int lane    = tid & 31;                   // wave32
  const int wave    = tid >> 5;                   // 8 waves cover COUT=128
  const int cobase  = wave << 4;
  const int halfSel = lane >> 4;                  // lanes 16-31: K groups {8-15,24-31}
  const int lm      = lane & 15;                  // M (A) / N (B) index

  const _Float16* __restrict__ wrow  = wh   + (cobase + lm) * CK;     // A: weight row M
  const _Float16* __restrict__ brow0 = cols + lm * LDSS;              // B tile 0: pixels 0..15
  const _Float16* __restrict__ brow1 = cols + (lm + 16) * LDSS;       // B tile 1: pixels 16..31

  v8f acc0 = {}, acc1 = {};
  #pragma unroll 4
  for (int kkb = 0; kkb < CK; kkb += 32) {
    v16h a, b0, b1;
    #pragma unroll
    for (int v = 0; v < 8; ++v) {
      // 16-bit A/B layout: v0..3 -> K {0..7}+8*halfSel, v4..7 -> K {16..23}+8*halfSel
      const int koff = kkb + ((v & 3) << 1) + (halfSel << 3) + ((v >> 2) << 4);
      a[2 * v]      = wrow[koff];
      a[2 * v + 1]  = wrow[koff + 1];
      b0[2 * v]     = brow0[koff];
      b0[2 * v + 1] = brow0[koff + 1];
      b1[2 * v]     = brow1[koff];
      b1[2 * v + 1] = brow1[koff + 1];
    }
    acc0 = __builtin_amdgcn_wmma_f32_16x16x32_f16(false, a, false, b0,
                                                  (short)0, acc0, false, false);
    acc1 = __builtin_amdgcn_wmma_f32_16x16x32_f16(false, a, false, b1,
                                                  (short)0, acc1, false, false);
  }

  // ---- write D: VGPR r -> M=r (lanes 0-15) / M=8+r (lanes 16-31), N=lm ----
  const int co0 = cobase + halfSel * 8;
  #pragma unroll
  for (int r = 0; r < 8; ++r) {
    const int co = co0 + r;
    const float bv = bias[co];
    float* orow = out + (((nb * COUT) + co) << 14) + pixBase;
    orow[lm]      = acc0[r] + bv;
    orow[16 + lm] = acc1[r] + bv;
  }
}

extern "C" void kernel_launch(void* const* d_in, const int* in_sizes, int n_in,
                              void* d_out, int out_size, void* d_ws, size_t ws_size,
                              hipStream_t stream) {
  const float* x      = (const float*)d_in[0];
  const float* offset = (const float*)d_in[1];
  const float* mask   = (const float*)d_in[2];
  const float* weight = (const float*)d_in[3];
  const float* bias   = (const float*)d_in[4];
  float* out          = (float*)d_out;
  _Float16* wh        = (_Float16*)d_ws;          // 128*1152*2 = 288 KB scratch

  const int wn = COUT * CK;
  hipLaunchKernelGGL(dcn_w2h, dim3((wn + 255) / 256), dim3(256), 0, stream,
                     weight, wh, wn);

  const int tiles = NB * HH * WW / TP;            // 1024 workgroups
  hipLaunchKernelGGL(dcn_main, dim3(tiles), dim3(256), 0, stream,
                     x, offset, mask, wh, bias, out);
}
